// SABlock_26972394619719
// MI455X (gfx1250) — compile-verified, matrix operand order checked by reading, not run on hardware
//
#include <hip/hip_runtime.h>
#include <hip/hip_bf16.h>

// ---------------------------------------------------------------------------
// Self-attention block for MI455X (gfx1250, wave32, WMMA).
// Pipeline:
//   1) convert fp32 -> bf16 (x, w_qkv, w_out)
//   2) QKV projection GEMM (bf16 WMMA, f32 accum), scatter to q[B,H,N,64],
//      k[B,H,N,64], vT[B,H,64,N]
//   3) fused flash-attention (online softmax), bf16 WMMA, output bf16 [B,N,768]
//   4) output projection GEMM + bias, fp32 result
// Tile staging uses CDNA5 async global->LDS DMA (ASYNCcnt) with LDS
// double-buffering: tile t+1 streams in while tile t is computed on; since
// async loads complete in order, s_wait_asynccnt <= (ops just issued)
// guarantees the previous tile has landed.
// ---------------------------------------------------------------------------

typedef __bf16 bf16;
typedef __attribute__((ext_vector_type(8)))  __bf16 bf16x8;
typedef __attribute__((ext_vector_type(16))) __bf16 v16bf;
typedef __attribute__((ext_vector_type(8)))  float  v8f;

union Frag {
    v16bf v;
    bf16x8 h[2];
};

#define HEADS    12
#define HEAD_DIM 64
#define HIDDEN   768
#define SEQ      4096
#define BATCH    2
#define ATT_SCALE 0.125f   // 64^-0.5

// CDNA5 async DMA: memory -> LDS, 16 bytes per lane, tracked by ASYNCcnt.
// LDS destination address = low 32 bits of the generic pointer (aperture rule).
__device__ __forceinline__ void async_lds_b128(const bf16* lds_dst, const bf16* gsrc) {
    unsigned int       ldsa = (unsigned int)(unsigned long long)(uintptr_t)lds_dst;
    unsigned long long ga   = (unsigned long long)(uintptr_t)gsrc;
    asm volatile("global_load_async_to_lds_b128 %0, %1, off"
                 :: "v"(ldsa), "v"(ga) : "memory");
}
__device__ __forceinline__ void wait_asynccnt0() {
    asm volatile("s_wait_asynccnt 0x0" ::: "memory");
}
__device__ __forceinline__ void wait_asynccnt2() {
    asm volatile("s_wait_asynccnt 0x2" ::: "memory");
}
__device__ __forceinline__ void wait_asynccnt3() {
    asm volatile("s_wait_asynccnt 0x3" ::: "memory");
}

// ---------------------------------------------------------------------------
// fp32 -> bf16 conversion (vectorized by 4)
// ---------------------------------------------------------------------------
__global__ void cvt_f32_to_bf16(const float4* __restrict__ src,
                                bf16* __restrict__ dst, int n4) {
    int i = blockIdx.x * blockDim.x + threadIdx.x;
    if (i < n4) {
        float4 v = src[i];
        dst[i * 4 + 0] = (bf16)v.x;
        dst[i * 4 + 1] = (bf16)v.y;
        dst[i * 4 + 2] = (bf16)v.z;
        dst[i * 4 + 3] = (bf16)v.w;
    }
}

// ---------------------------------------------------------------------------
// Tiled bf16 WMMA GEMM:  C[M,N] = A[M,K] * W[N,K]^T
// Block tile: 128(M) x 64(N), 8 waves arranged 4x2, each wave 32x32 (2x2 frags)
// Double-buffered async A/B tile prefetch (3 async ops per thread per tile).
// MODE 0: QKV epilogue (scatter q/k/vT, bf16)
// MODE 1: fp32 output + bias
// ---------------------------------------------------------------------------
template <int MODE>
__global__ __launch_bounds__(256)
void gemm_bf16_wmma(const bf16* __restrict__ A,
                    const bf16* __restrict__ W,
                    const float* __restrict__ bias,
                    float* __restrict__ outF,
                    bf16* __restrict__ qO,
                    bf16* __restrict__ kO,
                    bf16* __restrict__ vtO,
                    int M, int N, int K) {
    __shared__ bf16 Abuf[2][128 * 32];
    __shared__ bf16 Bbuf[2][64 * 32];

    const int tid   = threadIdx.x;
    const int wave  = tid >> 5;
    const int lane  = tid & 31;
    const int laneN = lane & 15;    // column (B/C/D) or row (A)
    const int laneH = lane >> 4;    // K-octet selector / row-half selector
    const int mw    = wave >> 1;    // 0..3
    const int nw    = wave & 1;     // 0..1
    const int m0    = blockIdx.x * 128;
    const int n0    = blockIdx.y * 64;

    auto issueTile = [&](int kt, int buf) {
        // A tile: 128 rows x 32 cols bf16 = 512 x 16B chunks, 2 per thread
#pragma unroll
        for (int it = 0; it < 2; ++it) {
            int c = tid + it * 256;
            int r = c >> 2, ch = c & 3;
            async_lds_b128(&Abuf[buf][r * 32 + ch * 8],
                           &A[(size_t)(m0 + r) * K + kt * 32 + ch * 8]);
        }
        // B tile (rows of W, i.e. B^T): 64 rows x 32 cols = 256 chunks
        int r = tid >> 2, ch = tid & 3;
        async_lds_b128(&Bbuf[buf][r * 32 + ch * 8],
                       &W[(size_t)(n0 + r) * K + kt * 32 + ch * 8]);
    };

    v8f acc[2][2] = {};
    const int KT = K >> 5;

    issueTile(0, 0);
#pragma unroll 2
    for (int kt = 0; kt < KT; ++kt) {
        const int cur = kt & 1;
        const bool pf = (kt + 1 < KT);
        if (pf) issueTile(kt + 1, cur ^ 1);   // stream next tile while computing
        if (pf) wait_asynccnt3(); else wait_asynccnt0();  // prev tile landed
        __syncthreads();

        Frag af[2], bf_[2];
#pragma unroll
        for (int i = 0; i < 2; ++i) {
            int ar = (mw * 32 + i * 16 + laneN) * 32;
            af[i].h[0] = *reinterpret_cast<const bf16x8*>(&Abuf[cur][ar + laneH * 8]);
            af[i].h[1] = *reinterpret_cast<const bf16x8*>(&Abuf[cur][ar + 16 + laneH * 8]);
            int br = (nw * 32 + i * 16 + laneN) * 32;
            bf_[i].h[0] = *reinterpret_cast<const bf16x8*>(&Bbuf[cur][br + laneH * 8]);
            bf_[i].h[1] = *reinterpret_cast<const bf16x8*>(&Bbuf[cur][br + 16 + laneH * 8]);
        }
#pragma unroll
        for (int i = 0; i < 2; ++i)
#pragma unroll
            for (int j = 0; j < 2; ++j)
                acc[i][j] = __builtin_amdgcn_wmma_f32_16x16x32_bf16(
                    false, af[i].v, false, bf_[j].v, (short)0, acc[i][j], false, false);
        __syncthreads();   // all waves done with buf[cur] before it is refilled
    }

    // Epilogue. C/D layout: lane = column N, VGPR r = row; lanes 16-31 hold rows 8-15.
#pragma unroll
    for (int i = 0; i < 2; ++i) {
#pragma unroll
        for (int j = 0; j < 2; ++j) {
            int oAbs = n0 + nw * 32 + j * 16 + laneN;
#pragma unroll
            for (int r = 0; r < 8; ++r) {
                int mAbs = m0 + mw * 32 + i * 16 + laneH * 8 + r;
                float v = acc[i][j][r];
                if (MODE == 0) {
                    // scatter to q/k/vT  (o = s*768 + h*64 + d)
                    int s   = oAbs / HIDDEN;
                    int rem = oAbs - s * HIDDEN;
                    int hh  = rem >> 6;
                    int d   = rem & 63;
                    int bb  = mAbs >> 12;          // token / 4096
                    int tok = mAbs & 4095;
                    size_t bh = (size_t)(bb * HEADS + hh);
                    bf16 bv = (bf16)v;
                    if (s == 0)      qO[(bh * SEQ + tok) * HEAD_DIM + d] = bv;
                    else if (s == 1) kO[(bh * SEQ + tok) * HEAD_DIM + d] = bv;
                    else             vtO[(bh * HEAD_DIM + d) * SEQ + tok] = bv;
                } else {
                    outF[(size_t)mAbs * N + oAbs] = v + bias[oAbs];
                }
            }
        }
    }
}

// ---------------------------------------------------------------------------
// Fused flash attention: grid (N/128, B*H), 256 threads (8 waves).
// Each wave owns 16 queries; 32-key K / V^T tiles double-buffered in LDS via
// async DMA (2 async ops per thread per tile).
// ---------------------------------------------------------------------------
__global__ __launch_bounds__(256)
void flash_attn_wmma(const bf16* __restrict__ Q,
                     const bf16* __restrict__ Kmat,
                     const bf16* __restrict__ Vt,
                     bf16* __restrict__ Ao) {
    __shared__ bf16 kbuf[2][32 * 64];       // [key][d]
    __shared__ bf16 vbuf[2][64 * 32];       // [d][key]  (V^T tile)
    __shared__ bf16 pbuf[8 * 16 * 32];      // per-wave P tile [m][n]

    const int tid   = threadIdx.x;
    const int wave  = tid >> 5;
    const int lane  = tid & 31;
    const int laneN = lane & 15;
    const int laneH = lane >> 4;

    const int bh = blockIdx.y;              // 0..23
    const int bb = bh / HEADS;
    const int hh = bh - bb * HEADS;
    const size_t baseQK = (size_t)bh * SEQ * HEAD_DIM;
    const size_t baseV  = (size_t)bh * HEAD_DIM * SEQ;
    const int q0 = blockIdx.x * 128 + wave * 16;

    auto issueKV = [&](int t, int buf) {
        {   // K tile: 32 rows x 64 d = 256 x 16B async chunks
            int r = tid >> 3, ch = tid & 7;
            async_lds_b128(&kbuf[buf][r * 64 + ch * 8],
                           &Kmat[baseQK + (size_t)(t * 32 + r) * HEAD_DIM + ch * 8]);
        }
        {   // V^T tile: 64 rows x 32 keys = 256 x 16B async chunks
            int r = tid >> 2, ch = tid & 3;
            async_lds_b128(&vbuf[buf][r * 32 + ch * 8],
                           &Vt[baseV + (size_t)r * SEQ + t * 32 + ch * 8]);
        }
    };

    // Q as A-matrix fragments (k-chunks 0..31 and 32..63), straight from HBM
    Frag qf[2];
    {
        const bf16* qrow = Q + baseQK + (size_t)(q0 + laneN) * HEAD_DIM;
        qf[0].h[0] = *reinterpret_cast<const bf16x8*>(qrow + laneH * 8);
        qf[0].h[1] = *reinterpret_cast<const bf16x8*>(qrow + 16 + laneH * 8);
        qf[1].h[0] = *reinterpret_cast<const bf16x8*>(qrow + 32 + laneH * 8);
        qf[1].h[1] = *reinterpret_cast<const bf16x8*>(qrow + 48 + laneH * 8);
    }

    float mrow[8], lrow[8];
    v8f oacc[4] = {};
#pragma unroll
    for (int r = 0; r < 8; ++r) { mrow[r] = -1e30f; lrow[r] = 0.f; }
    bf16* pw = pbuf + wave * (16 * 32);

    issueKV(0, 0);
#pragma unroll 2
    for (int t = 0; t < SEQ / 32; ++t) {
        const int cur = t & 1;
        const bool pf = (t + 1 < SEQ / 32);
        if (pf) issueKV(t + 1, cur ^ 1);    // stream next K/V while computing
        if (pf) wait_asynccnt2(); else wait_asynccnt0();
        __syncthreads();

        // S = Q K^T : 16x32 tile = two 16x16 accumulators, 4 WMMAs
        v8f sacc[2] = {};
#pragma unroll
        for (int kc = 0; kc < 2; ++kc) {
#pragma unroll
            for (int nh = 0; nh < 2; ++nh) {
                Frag bf_;
                const bf16* kr = &kbuf[cur][(nh * 16 + laneN) * 64 + kc * 32];
                bf_.h[0] = *reinterpret_cast<const bf16x8*>(kr + laneH * 8);
                bf_.h[1] = *reinterpret_cast<const bf16x8*>(kr + 16 + laneH * 8);
                sacc[nh] = __builtin_amdgcn_wmma_f32_16x16x32_bf16(
                    false, qf[kc].v, false, bf_.v, (short)0, sacc[nh], false, false);
            }
        }

        // Online softmax. Row statistics live replicated across the 16-lane half.
        // Fully unrolled: all v8f element indices must be compile-time constants
        // (dynamic indexing would force v_movrels + spills).
#pragma unroll
        for (int r = 0; r < 8; ++r) {
            float s0 = sacc[0][r] * ATT_SCALE;
            float s1 = sacc[1][r] * ATT_SCALE;
            float tm = fmaxf(s0, s1);
#pragma unroll
            for (int msk = 1; msk < 16; msk <<= 1)
                tm = fmaxf(tm, __shfl_xor(tm, msk, 32));
            float mn    = fmaxf(mrow[r], tm);
            float alpha = __expf(mrow[r] - mn);
            mrow[r] = mn;
            float p0 = __expf(s0 - mn);
            float p1 = __expf(s1 - mn);
            float ps = p0 + p1;
#pragma unroll
            for (int msk = 1; msk < 16; msk <<= 1)
                ps += __shfl_xor(ps, msk, 32);
            lrow[r] = lrow[r] * alpha + ps;
            int m = laneH * 8 + r;
            pw[m * 32 + laneN]      = (bf16)p0;   // C-layout -> row-major P tile
            pw[m * 32 + 16 + laneN] = (bf16)p1;
#pragma unroll
            for (int dh = 0; dh < 4; ++dh) oacc[dh][r] *= alpha;
        }

        // P re-read as an A-matrix fragment (intra-wave LDS round trip)
        Frag pfr;
        pfr.h[0] = *reinterpret_cast<const bf16x8*>(&pw[laneN * 32 + laneH * 8]);
        pfr.h[1] = *reinterpret_cast<const bf16x8*>(&pw[laneN * 32 + 16 + laneH * 8]);

        // O += P * V : 4 WMMAs (d split into 4 x 16)
#pragma unroll
        for (int dh = 0; dh < 4; ++dh) {
            Frag bf_;
            const bf16* vr = &vbuf[cur][(dh * 16 + laneN) * 32];
            bf_.h[0] = *reinterpret_cast<const bf16x8*>(vr + laneH * 8);
            bf_.h[1] = *reinterpret_cast<const bf16x8*>(vr + 16 + laneH * 8);
            oacc[dh] = __builtin_amdgcn_wmma_f32_16x16x32_bf16(
                false, pfr.v, false, bf_.v, (short)0, oacc[dh], false, false);
        }
        __syncthreads();   // all waves done with buf[cur] before it is refilled
    }

    // Normalize and store [B, N, H*64] bf16
#pragma unroll
    for (int dh = 0; dh < 4; ++dh) {
#pragma unroll
        for (int r = 0; r < 8; ++r) {
            int m   = laneH * 8 + r;
            int tok = q0 + m;
            int d   = dh * 16 + laneN;
            float v = oacc[dh][r] / lrow[r];
            Ao[((size_t)(bb * SEQ + tok)) * HIDDEN + hh * HEAD_DIM + d] = (bf16)v;
        }
    }
}

// ---------------------------------------------------------------------------
// Host launcher
// ---------------------------------------------------------------------------
extern "C" void kernel_launch(void* const* d_in, const int* in_sizes, int n_in,
                              void* d_out, int out_size, void* d_ws, size_t ws_size,
                              hipStream_t stream) {
    const float* x     = (const float*)d_in[0];   // [2,4096,768]
    const float* w_qkv = (const float*)d_in[1];   // [2304,768]
    const float* w_out = (const float*)d_in[2];   // [768,768]
    const float* b_out = (const float*)d_in[3];   // [768]
    float* out = (float*)d_out;                   // [2,4096,768]

    const int M    = BATCH * SEQ;                 // 8192 tokens
    const int Kdim = HIDDEN;                      // 768
    const int Nqkv = 3 * HIDDEN;                  // 2304

    char* ws = (char*)d_ws;
    size_t off = 0;
    auto carve = [&](size_t bytes) -> char* {
        char* p = ws + off;
        off += (bytes + 255) & ~(size_t)255;
        return p;
    };
    bf16* xb    = (bf16*)carve((size_t)M * Kdim * 2);
    bf16* wqkvb = (bf16*)carve((size_t)Nqkv * Kdim * 2);
    bf16* woutb = (bf16*)carve((size_t)HIDDEN * Kdim * 2);
    bf16* qb    = (bf16*)carve((size_t)BATCH * HEADS * SEQ * HEAD_DIM * 2);
    bf16* kb    = (bf16*)carve((size_t)BATCH * HEADS * SEQ * HEAD_DIM * 2);
    bf16* vtb   = (bf16*)carve((size_t)BATCH * HEADS * HEAD_DIM * SEQ * 2);
    bf16* aob   = (bf16*)carve((size_t)M * HIDDEN * 2);
    (void)ws_size; (void)n_in; (void)in_sizes; (void)out_size;

    // 1) convert inputs to bf16
    {
        int n4 = (M * Kdim) / 4;
        cvt_f32_to_bf16<<<(n4 + 255) / 256, 256, 0, stream>>>((const float4*)x, xb, n4);
        n4 = (Nqkv * Kdim) / 4;
        cvt_f32_to_bf16<<<(n4 + 255) / 256, 256, 0, stream>>>((const float4*)w_qkv, wqkvb, n4);
        n4 = (HIDDEN * Kdim) / 4;
        cvt_f32_to_bf16<<<(n4 + 255) / 256, 256, 0, stream>>>((const float4*)w_out, woutb, n4);
    }

    // 2) QKV projection + scatter to q / k / v^T
    {
        dim3 grid(M / 128, Nqkv / 64);
        gemm_bf16_wmma<0><<<grid, 256, 0, stream>>>(
            xb, wqkvb, nullptr, nullptr, qb, kb, vtb, M, Nqkv, Kdim);
    }

    // 3) fused flash attention
    {
        dim3 grid(SEQ / 128, BATCH * HEADS);
        flash_attn_wmma<<<grid, 256, 0, stream>>>(qb, kb, vtb, aob);
    }

    // 4) output projection + bias (fp32 result)
    {
        dim3 grid(M / 128, HIDDEN / 64);
        gemm_bf16_wmma<1><<<grid, 256, 0, stream>>>(
            aob, woutb, b_out, out, nullptr, nullptr, nullptr, M, HIDDEN, Kdim);
    }
}
